// cap_73083163508826
// MI455X (gfx1250) — matrix operand order repaired and checked by reading, not compiled, not run on hardware
//
#include <hip/hip_runtime.h>
#include <math.h>

typedef __attribute__((ext_vector_type(16))) __bf16 v16bf;
typedef __attribute__((ext_vector_type(8)))  float  v8f;

#define DEV static __device__ __forceinline__

// problem sizes (fixed by reference)
#define BB   16
#define NN   1024
#define DD   64
#define EE   16
#define HS   16
#define HT   64
#define KTT  192   // HS*T
#define BT   192   // B*T

DEV __bf16 f2bf(float f) { return static_cast<__bf16>(f); }  // native v_cvt (RTNE)
DEV float lrelu01(float x) { return x > 0.f ? x : 0.01f * x; }
DEV v8f vzero8() {
  v8f z;
#pragma unroll
  for (int i = 0; i < 8; i++) z[i] = 0.f;
  return z;
}
// A-operand K index (contiguous order k = 16*hf + s, matching B), pair base, p = vgpr-pair 0..7
DEV int akb(int hf, int p) { return hf * 16 + 2 * p; }

#define WMMA(A, Bv, C) \
  __builtin_amdgcn_wmma_f32_16x16x32_bf16(false, (A), false, (Bv), (short)0, (C), false, false)

// ---------------------------------------------------------------------------
// K0: convert W_p (64x64) to bf16 once
// ---------------------------------------------------------------------------
__global__ __launch_bounds__(256) void k_cvt_wp(const float* __restrict__ Wp,
                                                __bf16* __restrict__ Wpbf) {
  int idx = blockIdx.x * 256 + threadIdx.x;
  if (idx < DD * DD) Wpbf[idx] = f2bf(Wp[idx]);
}

// ---------------------------------------------------------------------------
// K1: Pcaps = squash(x @ W_p^T + b_p)  -> bf16 [BT*N, 64]
// ---------------------------------------------------------------------------
__global__ __launch_bounds__(256) void k_pcaps(const float* __restrict__ x,
                                               const __bf16* __restrict__ Wpbf,
                                               const float* __restrict__ bp,
                                               __bf16* __restrict__ Pbf, int totalRows) {
  const int lane = threadIdx.x & 31, wave = threadIdx.x >> 5;
  const int hf = lane >> 4, l16 = lane & 15;
  const int row0 = (blockIdx.x * 8 + wave) * 16;
  if (row0 >= totalRows) return;
  v8f acc[4];
#pragma unroll
  for (int j = 0; j < 4; j++) acc[j] = vzero8();
  const float* arow = x + (size_t)(row0 + l16) * DD;
#pragma unroll
  for (int ks = 0; ks < DD; ks += 32) {
    v16bf A;
#pragma unroll
    for (int p = 0; p < 8; p++) {
      int kb = akb(hf, p);
      A[2 * p]     = f2bf(arow[ks + kb]);
      A[2 * p + 1] = f2bf(arow[ks + kb + 1]);
    }
#pragma unroll
    for (int j = 0; j < 4; j++) {
      int o = j * 16 + l16;                    // B[k][o] = Wp[o*64 + k]
      const __bf16* bc = Wpbf + (size_t)o * DD + ks + hf * 16;
      v16bf Bv;
#pragma unroll
      for (int v = 0; v < 8; v++) {
        Bv[2 * v]     = bc[2 * v];
        Bv[2 * v + 1] = bc[2 * v + 1];
      }
      acc[j] = WMMA(A, Bv, acc[j]);
    }
  }
  // bias + per-row squash (row norm: reduce over 16 lanes of same half)
  float ps[8];
#pragma unroll
  for (int r = 0; r < 8; r++) ps[r] = 0.f;
#pragma unroll
  for (int j = 0; j < 4; j++) {
    float bv = bp[j * 16 + l16];
#pragma unroll
    for (int r = 0; r < 8; r++) {
      acc[j][r] += bv;
      ps[r] += acc[j][r] * acc[j][r];
    }
  }
#pragma unroll
  for (int r = 0; r < 8; r++) {
#pragma unroll
    for (int m = 1; m < 16; m <<= 1) ps[r] += __shfl_xor(ps[r], m, 16);
  }
#pragma unroll
  for (int r = 0; r < 8; r++) {
    float sn = ps[r];
    float sc = (sn / (1.f + sn)) / (sqrtf(sn) + 1e-8f);
    int row = row0 + hf * 8 + r;
#pragma unroll
    for (int j = 0; j < 4; j++)
      Pbf[(size_t)row * DD + j * 16 + l16] = f2bf(acc[j][r] * sc);
  }
}

// ---------------------------------------------------------------------------
// K2: dadj[bt,h,n] = sum_e teb[bt,e]*adj[e,h,n]
// ---------------------------------------------------------------------------
__global__ __launch_bounds__(256) void k_dadj(const float* __restrict__ teb,
                                              const float* __restrict__ adj,
                                              float* __restrict__ dadj) {
  int idx = blockIdx.x * 256 + threadIdx.x;
  if (idx >= BT * HS * NN) return;
  int n = idx & (NN - 1);
  int h = (idx >> 10) & (HS - 1);
  int bt = idx >> 14;
  float s = 0.f;
#pragma unroll
  for (int e = 0; e < EE; e++) s += teb[bt * EE + e] * adj[((size_t)e * HS + h) * NN + n];
  dadj[idx] = s;
}

// ---------------------------------------------------------------------------
// K3: softmax over h -> bf16 (always) + f32 (optional, for the c output).
// mode 0: softmax(b); 1: uniform 1/16; 2: softmax(b+add)
// ---------------------------------------------------------------------------
__global__ __launch_bounds__(256) void k_softmax_h(const float* __restrict__ bsrc,
                                                   const float* __restrict__ addsrc,
                                                   float* __restrict__ outf,
                                                   __bf16* __restrict__ outbf, int mode) {
  int idx = blockIdx.x * 256 + threadIdx.x;
  if (idx >= BT * NN) return;
  int n = idx & (NN - 1);
  int bt = idx >> 10;
  size_t base = (size_t)bt * HS * NN + n;
  if (mode == 1) {
    __bf16 u = f2bf(1.f / 16.f);
#pragma unroll
    for (int h = 0; h < HS; h++) {
      outbf[base + (size_t)h * NN] = u;
      if (outf) outf[base + (size_t)h * NN] = 1.f / 16.f;
    }
    return;
  }
  float v[HS];
  float mx = -3.4e38f;
#pragma unroll
  for (int h = 0; h < HS; h++) {
    float t = bsrc[base + (size_t)h * NN];
    if (mode == 2) t += addsrc[base + (size_t)h * NN];
    v[h] = t;
    mx = fmaxf(mx, t);
  }
  float s = 0.f;
#pragma unroll
  for (int h = 0; h < HS; h++) {
    v[h] = __expf(v[h] - mx);
    s += v[h];
  }
  float inv = 1.f / s;
#pragma unroll
  for (int h = 0; h < HS; h++) {
    float r = v[h] * inv;
    outbf[base + (size_t)h * NN] = f2bf(r);
    if (outf) outf[base + (size_t)h * NN] = r;
  }
}

// ---------------------------------------------------------------------------
// K4: per-bt GEMM  out[16,64] = A_bf[16,1024] @ P[1024,64], optional squash
// 8 waves split K, LDS reduction
// ---------------------------------------------------------------------------
__global__ __launch_bounds__(256) void k_capmat(const __bf16* __restrict__ Aall,
                                                const __bf16* __restrict__ P,
                                                float* __restrict__ out, int doSquash) {
  __shared__ float S[HS * DD];
  __shared__ float scl[HS];
  const int bt = blockIdx.x;
  const int lane = threadIdx.x & 31, wave = threadIdx.x >> 5;
  const int hf = lane >> 4, l16 = lane & 15;
  for (int i = threadIdx.x; i < HS * DD; i += 256) S[i] = 0.f;
  __syncthreads();
  const __bf16* Ab = Aall + (size_t)bt * HS * NN;
  const __bf16* Pb = P + (size_t)bt * NN * DD;
  v8f acc[4];
#pragma unroll
  for (int j = 0; j < 4; j++) acc[j] = vzero8();
  for (int kc = wave * 128; kc < wave * 128 + 128; kc += 32) {
    v16bf A;
#pragma unroll
    for (int p = 0; p < 8; p++) {
      int kb = akb(hf, p);
      A[2 * p]     = Ab[(size_t)l16 * NN + kc + kb];
      A[2 * p + 1] = Ab[(size_t)l16 * NN + kc + kb + 1];
    }
#pragma unroll
    for (int j = 0; j < 4; j++) {
      int d = j * 16 + l16;
      int kb2 = kc + hf * 16;
      v16bf Bv;
#pragma unroll
      for (int v = 0; v < 8; v++) {
        Bv[2 * v]     = Pb[(size_t)(kb2 + 2 * v) * DD + d];
        Bv[2 * v + 1] = Pb[(size_t)(kb2 + 2 * v + 1) * DD + d];
      }
      acc[j] = WMMA(A, Bv, acc[j]);
    }
  }
#pragma unroll
  for (int j = 0; j < 4; j++)
#pragma unroll
    for (int r = 0; r < 8; r++)
      atomicAdd(&S[(hf * 8 + r) * DD + j * 16 + l16], acc[j][r]);
  __syncthreads();
  if (threadIdx.x < HS) {
    float sn = 0.f;
    for (int d = 0; d < DD; d++) {
      float t = S[threadIdx.x * DD + d];
      sn += t * t;
    }
    scl[threadIdx.x] = doSquash ? (sn / (1.f + sn)) / (sqrtf(sn) + 1e-8f) : 1.f;
  }
  __syncthreads();
  float* ob = out + (size_t)bt * HS * DD;
  for (int i = threadIdx.x; i < HS * DD; i += 256) ob[i] = S[i] * scl[i >> 6];
}

// ---------------------------------------------------------------------------
// K5: one routing iteration: S=c@P ; v=squash(sq1*S) ; b (+)= v @ P^T
// ---------------------------------------------------------------------------
__global__ __launch_bounds__(256) void k_route(const __bf16* __restrict__ c,
                                               const __bf16* __restrict__ P,
                                               const float* __restrict__ sq1,
                                               float* __restrict__ b, int addOld) {
  __shared__ float S[HS * DD];
  __shared__ float scl[HS];
  const int bt = blockIdx.x;
  const int lane = threadIdx.x & 31, wave = threadIdx.x >> 5;
  const int hf = lane >> 4, l16 = lane & 15;
  for (int i = threadIdx.x; i < HS * DD; i += 256) S[i] = 0.f;
  __syncthreads();
  const __bf16* Ab = c + (size_t)bt * HS * NN;
  const __bf16* Pb = P + (size_t)bt * NN * DD;
  {  // stage 1: S = c @ P  (split-K over 8 waves)
    v8f acc[4];
#pragma unroll
    for (int j = 0; j < 4; j++) acc[j] = vzero8();
    for (int kc = wave * 128; kc < wave * 128 + 128; kc += 32) {
      v16bf A;
#pragma unroll
      for (int p = 0; p < 8; p++) {
        int kb = akb(hf, p);
        A[2 * p]     = Ab[(size_t)l16 * NN + kc + kb];
        A[2 * p + 1] = Ab[(size_t)l16 * NN + kc + kb + 1];
      }
#pragma unroll
      for (int j = 0; j < 4; j++) {
        int d = j * 16 + l16;
        int kb2 = kc + hf * 16;
        v16bf Bv;
#pragma unroll
        for (int v = 0; v < 8; v++) {
          Bv[2 * v]     = Pb[(size_t)(kb2 + 2 * v) * DD + d];
          Bv[2 * v + 1] = Pb[(size_t)(kb2 + 2 * v + 1) * DD + d];
        }
        acc[j] = WMMA(A, Bv, acc[j]);
      }
    }
#pragma unroll
    for (int j = 0; j < 4; j++)
#pragma unroll
      for (int r = 0; r < 8; r++)
        atomicAdd(&S[(hf * 8 + r) * DD + j * 16 + l16], acc[j][r]);
  }
  __syncthreads();
  const float* sq = sq1 + (size_t)bt * HS * DD;
  for (int i = threadIdx.x; i < HS * DD; i += 256) S[i] *= sq[i];
  __syncthreads();
  if (threadIdx.x < HS) {
    float sn = 0.f;
    for (int d = 0; d < DD; d++) {
      float t = S[threadIdx.x * DD + d];
      sn += t * t;
    }
    scl[threadIdx.x] = (sn / (1.f + sn)) / (sqrtf(sn) + 1e-8f);
  }
  __syncthreads();
  // stage 2: b[h,n] (+)= sum_d v[h,d]*P[n,d] ; A = v from LDS, B = P rows
  v16bf Av[2];
#pragma unroll
  for (int ks = 0; ks < 2; ks++) {
#pragma unroll
    for (int p = 0; p < 8; p++) {
      int kb = akb(hf, p);
      Av[ks][2 * p]     = f2bf(S[l16 * DD + ks * 32 + kb] * scl[l16]);
      Av[ks][2 * p + 1] = f2bf(S[l16 * DD + ks * 32 + kb + 1] * scl[l16]);
    }
  }
  float* bb = b + (size_t)bt * HS * NN;
  for (int t = 0; t < 8; t++) {
    int n0 = wave * 128 + t * 16;
    int n = n0 + l16;
    v8f acc = vzero8();
#pragma unroll
    for (int ks = 0; ks < 2; ks++) {
      const __bf16* pr = Pb + (size_t)n * DD + ks * 32 + hf * 16;  // P[n][k] contiguous
      v16bf Bv;
#pragma unroll
      for (int v = 0; v < 8; v++) {
        Bv[2 * v]     = pr[2 * v];
        Bv[2 * v + 1] = pr[2 * v + 1];
      }
      acc = WMMA(Av[ks], Bv, acc);
    }
#pragma unroll
    for (int r = 0; r < 8; r++) {
      size_t oi = (size_t)(hf * 8 + r) * NN + n0 + l16;
      float old = addOld ? bb[oi] : 0.f;
      bb[oi] = old + acc[r];
    }
  }
}

// ---------------------------------------------------------------------------
// K8: dyn[b,h,k] = sum_e time_eb[b,e]*t_adj[e,h,k]
// ---------------------------------------------------------------------------
__global__ __launch_bounds__(256) void k_dyn(const float* __restrict__ timeeb,
                                             const float* __restrict__ tadj,
                                             float* __restrict__ dyn) {
  int idx = blockIdx.x * 256 + threadIdx.x;
  if (idx >= BB * HT * KTT) return;
  int k = idx % KTT;
  int h = (idx / KTT) & (HT - 1);
  int b = idx / (KTT * HT);
  float s = 0.f;
#pragma unroll
  for (int e = 0; e < EE; e++) s += timeeb[b * EE + e] * tadj[((size_t)e * HT + h) * KTT + k];
  dyn[idx] = s;
}

// ---------------------------------------------------------------------------
// K9a: tem[b,h,d] = lrelu(sum_k dyn[b,h,k]*(s[b,k,d]+mask(k)))
// ---------------------------------------------------------------------------
__global__ __launch_bounds__(256) void k_tem(const float* __restrict__ dyn,
                                             const float* __restrict__ s,
                                             float* __restrict__ tem) {
  int idx = blockIdx.x * 256 + threadIdx.x;
  if (idx >= BB * HT * DD) return;
  int d = idx & (DD - 1);
  int h = (idx >> 6) & (HT - 1);
  int b = idx >> 12;
  const float* dr = dyn + ((size_t)b * HT + h) * KTT;
  const float* sb = s + (size_t)b * KTT * DD;
  float a = 0.f;
  for (int k = 0; k < KTT; k++)
    a += dr[k] * (sb[(size_t)k * DD + d] + (float)(k / 16 + 1) * (1.f / 12.f));
  tem[idx] = lrelu01(a);
}

// ---------------------------------------------------------------------------
// K9b: ret = lrelu(dyn^T @ tem)+s ; v2 = squash(ret) per row (64 thr / row)
// ---------------------------------------------------------------------------
__global__ __launch_bounds__(64) void k_retv2(const float* __restrict__ dyn,
                                              const float* __restrict__ tem,
                                              const float* __restrict__ s,
                                              float* __restrict__ v2) {
  __shared__ float red[DD];
  int b = blockIdx.x / KTT, k = blockIdx.x % KTT, d = threadIdx.x;
  float a = 0.f;
  for (int h = 0; h < HT; h++)
    a += dyn[((size_t)b * HT + h) * KTT + k] * tem[((size_t)b * HT + h) * DD + d];
  float ret = lrelu01(a) + s[((size_t)b * KTT + k) * DD + d];
  red[d] = ret * ret;
  __syncthreads();
  for (int off = 32; off >= 1; off >>= 1) {
    if (d < off) red[d] += red[d + off];
    __syncthreads();
  }
  float sn = red[0];
  float sc = (sn / (1.f + sn)) / (sqrtf(sn) + 1e-8f);
  v2[((size_t)b * KTT + k) * DD + d] = ret * sc;
}

// ---------------------------------------------------------------------------
// K10: recon[n,d] = sum_h c[h,n]*v2[h,d]  (K=16, zero-padded to 32)  -> bf16
// ---------------------------------------------------------------------------
__global__ __launch_bounds__(256) void k_recon(const __bf16* __restrict__ c,
                                               const float* __restrict__ v2,
                                               __bf16* __restrict__ recon) {
  const int bt = blockIdx.x;
  const int lane = threadIdx.x & 31, wave = threadIdx.x >> 5;
  const int hf = lane >> 4, l16 = lane & 15;
  const __bf16* cb = c + (size_t)bt * HS * NN;
  const float* vb = v2 + (size_t)bt * HS * DD;
  const __bf16 zb = f2bf(0.f);
  v16bf Bv[4];
#pragma unroll
  for (int j = 0; j < 4; j++) {
    int d = j * 16 + l16;
#pragma unroll
    for (int v = 0; v < 8; v++) {
      int k = hf * 16 + 2 * v;
      Bv[j][2 * v]     = (k < 16) ? f2bf(vb[(size_t)k * DD + d]) : zb;
      Bv[j][2 * v + 1] = (k + 1 < 16) ? f2bf(vb[(size_t)(k + 1) * DD + d]) : zb;
    }
  }
  for (int t = 0; t < 8; t++) {
    int n0 = wave * 128 + t * 16;
    int n = n0 + l16;
    v16bf Av;
#pragma unroll
    for (int p = 0; p < 8; p++) {
      int kb = akb(hf, p);
      Av[2 * p]     = (kb < 16) ? cb[(size_t)kb * NN + n] : zb;
      Av[2 * p + 1] = (kb + 1 < 16) ? cb[(size_t)(kb + 1) * NN + n] : zb;
    }
#pragma unroll
    for (int j = 0; j < 4; j++) {
      v8f acc = vzero8();
      acc = WMMA(Av, Bv[j], acc);
#pragma unroll
      for (int r = 0; r < 8; r++)
        recon[((size_t)bt * NN + n0 + hf * 8 + r) * DD + j * 16 + l16] = f2bf(acc[r]);
    }
  }
}

// ---------------------------------------------------------------------------
// K11a/b: w_sp (bf16) and b_sp (f32)
// ---------------------------------------------------------------------------
__global__ __launch_bounds__(256) void k_wsp(const float* __restrict__ nemb,
                                             const float* __restrict__ wspa,
                                             __bf16* __restrict__ wsp) {
  int idx = blockIdx.x * 256 + threadIdx.x;
  if (idx >= NN * DD * DD) return;
  int o = idx & (DD - 1);
  int i = (idx >> 6) & (DD - 1);
  int n = idx >> 12;
  float s = 0.f;
#pragma unroll
  for (int e = 0; e < EE; e++) s += nemb[n * EE + e] * wspa[((size_t)e * DD + i) * DD + o];
  wsp[idx] = f2bf(s);
}

__global__ __launch_bounds__(256) void k_bsp(const float* __restrict__ nemb,
                                             const float* __restrict__ bspa,
                                             float* __restrict__ bsp) {
  int idx = blockIdx.x * 256 + threadIdx.x;
  if (idx >= NN * DD) return;
  int o = idx & (DD - 1);
  int n = idx >> 6;
  float s = 0.f;
#pragma unroll
  for (int e = 0; e < EE; e++) s += nemb[n * EE + e] * bspa[e * DD + o];
  bsp[idx] = s;
}

// ---------------------------------------------------------------------------
// K12: per-n GEMM  out[bt,n,o] = lrelu( recon[bt,n,:]@w_sp[n] + b_sp[n,o] + x )
// ---------------------------------------------------------------------------
__global__ __launch_bounds__(128) void k_final(const __bf16* __restrict__ recon,
                                               const __bf16* __restrict__ wsp,
                                               const float* __restrict__ bsp,
                                               const float* __restrict__ x,
                                               float* __restrict__ out) {
  const int n = blockIdx.x;
  const int lane = threadIdx.x & 31, wave = threadIdx.x >> 5;  // 4 waves
  const int hf = lane >> 4, l16 = lane & 15;
  const __bf16* wn = wsp + (size_t)n * DD * DD;
  for (int tt = wave; tt < 12; tt += 4) {
    int r0 = tt * 16;  // bt row base
    v8f acc[4];
#pragma unroll
    for (int j = 0; j < 4; j++) acc[j] = vzero8();
#pragma unroll
    for (int ks = 0; ks < DD; ks += 32) {
      const __bf16* ar = recon + ((size_t)(r0 + l16) * NN + n) * DD + ks;
      v16bf Av;
#pragma unroll
      for (int p = 0; p < 8; p++) {
        int kb = akb(hf, p);
        Av[2 * p]     = ar[kb];
        Av[2 * p + 1] = ar[kb + 1];
      }
#pragma unroll
      for (int j = 0; j < 4; j++) {
        int o = j * 16 + l16;
        const __bf16* br = wn + (size_t)(ks + hf * 16) * DD + o;
        v16bf Bv;
#pragma unroll
        for (int v = 0; v < 8; v++) {
          Bv[2 * v]     = br[(size_t)(2 * v) * DD];
          Bv[2 * v + 1] = br[(size_t)(2 * v + 1) * DD];
        }
        acc[j] = WMMA(Av, Bv, acc[j]);
      }
    }
#pragma unroll
    for (int j = 0; j < 4; j++) {
      int o = j * 16 + l16;
      float bias = bsp[(size_t)n * DD + o];
#pragma unroll
      for (int r = 0; r < 8; r++) {
        size_t oi = ((size_t)(r0 + hf * 8 + r) * NN + n) * DD + o;
        out[oi] = lrelu01(acc[j][r] + bias + x[oi]);
      }
    }
  }
}

// ---------------------------------------------------------------------------
extern "C" void kernel_launch(void* const* d_in, const int* in_sizes, int n_in,
                              void* d_out, int out_size, void* d_ws, size_t ws_size,
                              hipStream_t stream) {
  (void)in_sizes; (void)n_in; (void)out_size; (void)ws_size;
  const float* x      = (const float*)d_in[0];
  const float* nemb   = (const float*)d_in[1];
  const float* timeeb = (const float*)d_in[2];
  const float* teb    = (const float*)d_in[3];
  const float* Wp     = (const float*)d_in[4];
  const float* bp     = (const float*)d_in[5];
  const float* tadj   = (const float*)d_in[6];
  const float* adj    = (const float*)d_in[7];
  const float* wspa   = (const float*)d_in[8];
  const float* bspa   = (const float*)d_in[9];

  float* out0    = (float*)d_out;                      // [BT,N,D]
  float* c_out   = out0 + (size_t)BT * NN * DD;        // [BT,HS,N]
  float* dyn_out = c_out + (size_t)BT * HS * NN;       // [B,HT,KTT]

  char* w = (char*)d_ws;
  __bf16* Pbf    = (__bf16*)w;  w += (size_t)BT * NN * DD * 2;
  float* dadj    = (float*)w;   w += (size_t)BT * HS * NN * 4;
  float* brt     = (float*)w;   w += (size_t)BT * HS * NN * 4;
  __bf16* sda_bf = (__bf16*)w;  w += (size_t)BT * HS * NN * 2;
  __bf16* crt_bf = (__bf16*)w;  w += (size_t)BT * HS * NN * 2;
  __bf16* cfn_bf = (__bf16*)w;  w += (size_t)BT * HS * NN * 2;
  float* sq1     = (float*)w;   w += (size_t)BT * HS * DD * 4;
  float* sbuf    = (float*)w;   w += (size_t)BT * HS * DD * 4;
  float* v2buf   = (float*)w;   w += (size_t)BT * HS * DD * 4;
  float* tem     = (float*)w;   w += (size_t)BB * HT * DD * 4;
  __bf16* recon  = (__bf16*)w;  w += (size_t)BT * NN * DD * 2;
  __bf16* wsp    = (__bf16*)w;  w += (size_t)NN * DD * DD * 2;
  float* bsp     = (float*)w;   w += (size_t)NN * DD * 4;
  __bf16* wpbf   = (__bf16*)w;  w += (size_t)DD * DD * 2;

  k_cvt_wp<<<(DD * DD + 255) / 256, 256, 0, stream>>>(Wp, wpbf);
  k_pcaps<<<1536, 256, 0, stream>>>(x, wpbf, bp, Pbf, BT * NN);
  k_dadj<<<(BT * HS * NN) / 256, 256, 0, stream>>>(teb, adj, dadj);
  k_softmax_h<<<(BT * NN) / 256, 256, 0, stream>>>(dadj, nullptr, nullptr, sda_bf, 0);
  k_capmat<<<BT, 256, 0, stream>>>(sda_bf, Pbf, sq1, 1);
  for (int it = 0; it < 3; it++) {
    k_softmax_h<<<(BT * NN) / 256, 256, 0, stream>>>(brt, nullptr, nullptr, crt_bf,
                                                     it == 0 ? 1 : 0);
    k_route<<<BT, 256, 0, stream>>>(crt_bf, Pbf, sq1, brt, it == 0 ? 0 : 1);
  }
  k_softmax_h<<<(BT * NN) / 256, 256, 0, stream>>>(brt, dadj, c_out, cfn_bf, 2);
  k_capmat<<<BT, 256, 0, stream>>>(cfn_bf, Pbf, sbuf, 0);
  k_dyn<<<(BB * HT * KTT) / 256, 256, 0, stream>>>(timeeb, tadj, dyn_out);
  k_tem<<<(BB * HT * DD) / 256, 256, 0, stream>>>(dyn_out, sbuf, tem);
  k_retv2<<<BB * KTT, 64, 0, stream>>>(dyn_out, tem, sbuf, v2buf);
  k_recon<<<BT, 256, 0, stream>>>(cfn_bf, v2buf, recon);
  k_wsp<<<(NN * DD * DD) / 256, 256, 0, stream>>>(nemb, wspa, wsp);
  k_bsp<<<(NN * DD) / 256, 256, 0, stream>>>(nemb, bspa, bsp);
  k_final<<<NN, 128, 0, stream>>>(recon, wsp, bsp, x, out0);
}
